// TorchRNNModel_18897856102917
// MI455X (gfx1250) — compile-verified, matrix operand order checked by reading, not run on hardware
//
#include <hip/hip_runtime.h>
#include <hip/hip_bf16.h>
#include <stdint.h>

#define B_  1024
#define S_  256
#define D_  128
#define H2_ 5
#define EPS_ 1e-5f

// ---------- CDNA5 WMMA types ----------
typedef __attribute__((ext_vector_type(16))) __bf16 v16bf;
typedef __attribute__((ext_vector_type(8)))  __bf16 v8bf;
typedef __attribute__((ext_vector_type(8)))  float  v8f;
typedef __attribute__((ext_vector_type(4)))  unsigned int u32x4;
typedef __attribute__((ext_vector_type(8)))  int i32x8;
typedef __attribute__((ext_vector_type(4)))  int i32x4;

// Padded LDS row strides (bank-conflict-free for ds_read_b128 column patterns)
#define WSTRIDE 136   // bf16 elements: 128 + 8  (= TDM pad of 4 dwords / 64-dword interval)
#define HSTRIDE 136
#define FSTRIDE 132   // f32 staging rows

#if defined(__gfx1250__) && __has_builtin(__builtin_amdgcn_tensor_load_to_lds) && __has_builtin(__builtin_amdgcn_s_wait_tensorcnt)
#define USE_TDM 1
#else
#define USE_TDM 0
#endif

// Build v16bf A/B operand per ISA 7.12.2 16-bit layout:
//  lanes 0-15 : K = kbase..kbase+7  then kbase+16..kbase+23
//  lanes 16-31: K = kbase+8..+15    then kbase+24..+31
// caller passes pointer already offset by (row*stride + ktile*32 + hi*8)
__device__ inline v16bf load16(const __bf16* p) {
  v8bf lo = *(const v8bf*)p;
  v8bf hh = *(const v8bf*)(p + 16);
  return __builtin_shufflevector(lo, hh, 0,1,2,3,4,5,6,7,8,9,10,11,12,13,14,15);
}

#if USE_TDM
// Issue a 2D TDM load of rows x cols bf16 elements (contiguous in global) into
// LDS with 4-dword padding every 64 dwords (row stride becomes 136 elements).
__device__ inline void tdm_issue(__bf16* lds_dst, const __bf16* gsrc,
                                 uint32_t rows, uint32_t cols) {
  uint64_t ga = (uint64_t)(uintptr_t)gsrc;
  uint32_t lo = (uint32_t)(uintptr_t)lds_dst;      // low 32 bits of generic == LDS offset
  u32x4 g0;
  g0[0] = 1u;                                       // count=1, user descriptor
  g0[1] = lo;                                       // lds_addr
  g0[2] = (uint32_t)ga;                             // global_addr[31:0]
  g0[3] = (uint32_t)((ga >> 32) & 0x1FFFFFFu) | (2u << 30);  // addr[56:32] | type=2
  uint64_t stride1 = (uint64_t)rows * cols;
  i32x8 g1;
  // data_size=1 (2B) | pad_enable | pad_interval=5 (64 dw) | pad_amount=3 (4 dw)
  g1[0] = (int)((1u << 16) | (1u << 20) | (5u << 22) | (3u << 25));
  g1[1] = (int)((cols & 0xFFFFu) << 16);                              // tensor_dim0 lo
  g1[2] = (int)(((cols >> 16) & 0xFFFFu) | ((rows & 0xFFFFu) << 16)); // dim0 hi | dim1 lo
  g1[3] = (int)(((rows >> 16) & 0xFFFFu) | ((cols & 0xFFFFu) << 16)); // dim1 hi | tile_dim0
  g1[4] = (int)(rows & 0xFFFFu);                                      // tile_dim1 (tile_dim2=0)
  g1[5] = (int)cols;                                                  // tensor_dim0_stride lo32
  g1[6] = (int)((stride1 & 0xFFFFu) << 16);                           // dim1_stride lo16
  g1[7] = (int)((stride1 >> 16) & 0xFFFFFFFFu);                       // dim1_stride hi
  i32x4 gz = {0, 0, 0, 0};
#if __clang_major__ >= 23
  i32x8 gz8 = {0, 0, 0, 0, 0, 0, 0, 0};
  __builtin_amdgcn_tensor_load_to_lds(g0, g1, gz, gz, gz8, 0);
#else
  __builtin_amdgcn_tensor_load_to_lds(g0, g1, gz, gz, 0);
#endif
}
#endif

__device__ inline void load_weights_lds(__bf16* dst, const __bf16* src) {
#if USE_TDM
  if (threadIdx.x < 32) tdm_issue(dst, src, 128u, 128u);
#else
  for (int idx = threadIdx.x; idx < 128 * 128; idx += blockDim.x)
    dst[(idx >> 7) * WSTRIDE + (idx & 127)] = src[idx];
#endif
}

__device__ inline void weights_wait() {
#if USE_TDM
  if (threadIdx.x < 32) __builtin_amdgcn_s_wait_tensorcnt(0);
#endif
  __syncthreads();
}

// ---------------- Kernel 0: weight prep (f32 -> bf16, fused biases) ----------------
__global__ void prep_kernel(const float* __restrict__ Wih1, const float* __restrict__ Whh1,
                            const float* __restrict__ bih1, const float* __restrict__ bhh1,
                            const float* __restrict__ bih2, const float* __restrict__ bhh2,
                            __bf16* wih1_bf, __bf16* whh1_bf,
                            float* bias1, float* bias2) {
  int tid = blockIdx.x * blockDim.x + threadIdx.x;
  int nthreads = blockDim.x * gridDim.x;
  for (int i = tid; i < D_ * D_; i += nthreads) {
    wih1_bf[i] = (__bf16)Wih1[i];
    whh1_bf[i] = (__bf16)Whh1[i];
  }
  if (tid < D_)  bias1[tid] = bih1[tid] + bhh1[tid];
  if (tid < H2_) bias2[tid] = bih2[tid] + bhh2[tid];
}

// ---------------- Kernel 1: embedding gather + xproj1 GEMM ----------------
// Time-major output: xout[(t*B + b)*D + d] = bf16( emb[x[b,t]] @ Wih1^T + bias1 )
// Block: 128 threads (4 waves), 64 rows per block, each wave: 16 M-rows x 128 N.
__global__ void __launch_bounds__(128)
xproj_kernel(const int* __restrict__ x, const float* __restrict__ emb,
             const __bf16* __restrict__ wih1_bf, const float* __restrict__ bias1,
             __bf16* __restrict__ xout) {
  __shared__ __align__(16) __bf16 Wsh[D_ * WSTRIDE];
  __shared__ __align__(16) __bf16 Etile[64 * HSTRIDE];
  __shared__ float b1s[D_];

  const int tid = threadIdx.x;
  const int rbase = blockIdx.x * 64;          // row index r = t*B + b

  load_weights_lds(Wsh, wih1_bf);
  if (tid < D_) b1s[tid] = bias1[tid];

  // gather + f32->bf16 convert: thread handles half a row (64 elements)
  {
    int row = tid >> 1, half = tid & 1;
    int r = rbase + row;
    int t = r >> 10;                          // r / B  (B=1024)
    int b = r & (B_ - 1);                     // r % B
    int token = x[(size_t)b * S_ + t];
    const float* src = emb + (size_t)token * D_ + half * 64;
    __bf16* dst = Etile + row * HSTRIDE + half * 64;
    for (int k = 0; k < 64; k += 8) {
      float4 f0 = *(const float4*)(src + k);
      float4 f1 = *(const float4*)(src + k + 4);
      v8bf v;
      v[0] = (__bf16)f0.x; v[1] = (__bf16)f0.y; v[2] = (__bf16)f0.z; v[3] = (__bf16)f0.w;
      v[4] = (__bf16)f1.x; v[5] = (__bf16)f1.y; v[6] = (__bf16)f1.z; v[7] = (__bf16)f1.w;
      *(v8bf*)(dst + k) = v;
    }
  }
  weights_wait();

  const int lane = tid & 31, wv = tid >> 5;
  const int rsel = lane & 15, hi = lane >> 4;

  // A tiles for this wave's 16 rows (4 k-chunks of 32)
  v16bf a[4];
  for (int kt = 0; kt < 4; ++kt)
    a[kt] = load16(&Etile[(wv * 16 + rsel) * HSTRIDE + kt * 32 + hi * 8]);

  for (int nt = 0; nt < 8; ++nt) {
    v8f c = {};
    for (int kt = 0; kt < 4; ++kt) {
      v16bf bm = load16(&Wsh[(nt * 16 + rsel) * WSTRIDE + kt * 32 + hi * 8]);
      c = __builtin_amdgcn_wmma_f32_16x16x32_bf16(false, a[kt], false, bm,
                                                  (short)0, c, false, false);
    }
    float bb = b1s[nt * 16 + rsel];
    // C/D layout: c[i] -> (m = 8*hi + i, n = rsel) within the 16x16 tile
    for (int i = 0; i < 8; ++i) {
      size_t r = (size_t)(rbase + wv * 16 + hi * 8 + i);
      xout[r * D_ + nt * 16 + rsel] = (__bf16)(c[i] + bb);
    }
  }
}

// ---------------- Kernel 2: fused RNN1 + LN1 + RNN2 + LN2 + pool + softmax ----------------
// 64 blocks, each owns 16 batch rows; 128 threads (4 waves); serial over S=256.
__global__ void __launch_bounds__(128)
rnn_kernel(const __bf16* __restrict__ xproj, const __bf16* __restrict__ whh_bf,
           const float* __restrict__ g1, const float* __restrict__ be1,
           const float* __restrict__ Wih2, const float* __restrict__ bias2,
           const float* __restrict__ Whh2, const float* __restrict__ g2,
           const float* __restrict__ be2, float* __restrict__ out) {
  __shared__ __align__(16) __bf16 Wsh[D_ * WSTRIDE];   // W_hh1, TDM-staged
  __shared__ __align__(16) __bf16 Hs[16 * HSTRIDE];    // recurrent state (bf16 A operand)
  __shared__ float Hnew[16 * FSTRIDE];                 // f32 h / LN(z) staging
  __shared__ float red_s[16][8], red_q[16][8];
  __shared__ float mrs[16][2];
  __shared__ float Wih2s[H2_ * D_];
  __shared__ float Whh2s[H2_ * H2_], b2s[H2_], g2s[H2_], be2s[H2_];
  __shared__ float g1s[D_], be1s[D_];
  __shared__ float h2[16 * H2_], h2n[16 * H2_], pooled[16 * H2_];

  const int tid = threadIdx.x;
  const int bbase = blockIdx.x * 16;

  load_weights_lds(Wsh, whh_bf);
  for (int i = tid; i < 16 * HSTRIDE; i += 128) Hs[i] = (__bf16)0.f;
  for (int i = tid; i < H2_ * D_; i += 128) Wih2s[i] = Wih2[i];
  if (tid < D_) { g1s[tid] = g1[tid]; be1s[tid] = be1[tid]; }
  if (tid < H2_ * H2_) Whh2s[tid] = Whh2[tid];
  if (tid < H2_) { b2s[tid] = bias2[tid]; g2s[tid] = g2[tid]; be2s[tid] = be2[tid]; }
  for (int i = tid; i < 16 * H2_; i += 128) { h2[i] = 0.f; pooled[i] = 0.f; }
  weights_wait();

  const int lane = tid & 31, wv = tid >> 5;
  const int rsel = lane & 15, hi = lane >> 4;

  for (int t = 0; t < S_; ++t) {
    // ---- phase 1: h = tanh(xproj_t + H @ W_hh^T), WMMA; wave wv owns n-tiles 2wv, 2wv+1
    v16bf a[4];
    for (int kt = 0; kt < 4; ++kt)
      a[kt] = load16(&Hs[rsel * HSTRIDE + kt * 32 + hi * 8]);

    float hval[2][8];
    for (int q = 0; q < 2; ++q) {
      int nt = wv * 2 + q;
      v8f c;
      const __bf16* xp = xproj + ((size_t)t * B_ + bbase) * D_ + nt * 16 + rsel;
      for (int i = 0; i < 8; ++i) c[i] = (float)xp[(size_t)(hi * 8 + i) * D_];
      for (int kt = 0; kt < 4; ++kt) {
        v16bf bm = load16(&Wsh[(nt * 16 + rsel) * WSTRIDE + kt * 32 + hi * 8]);
        c = __builtin_amdgcn_wmma_f32_16x16x32_bf16(false, a[kt], false, bm,
                                                    (short)0, c, false, false);
      }
      for (int i = 0; i < 8; ++i) hval[q][i] = tanhf(c[i]);
    }
    __syncthreads();   // all waves finished reading Hs

    // ---- phase 2: commit new state (bf16) + f32 staging for LN
    for (int q = 0; q < 2; ++q) {
      int n = (wv * 2 + q) * 16 + rsel;
      for (int i = 0; i < 8; ++i) {
        int m = hi * 8 + i;
        Hnew[m * FSTRIDE + n] = hval[q][i];
        Hs[m * HSTRIDE + n] = (__bf16)hval[q][i];
      }
    }
    __syncthreads();

    // ---- phase 3: LayerNorm over D=128, per row (16 rows x 8 segments)
    {
      int row = tid >> 3, seg = tid & 7;
      float s = 0.f, qq = 0.f;
      for (int k = 0; k < 16; ++k) {
        float v = Hnew[row * FSTRIDE + seg * 16 + k];
        s += v; qq += v * v;
      }
      red_s[row][seg] = s; red_q[row][seg] = qq;
    }
    __syncthreads();
    if (tid < 16) {
      float s = 0.f, qq = 0.f;
      for (int k = 0; k < 8; ++k) { s += red_s[tid][k]; qq += red_q[tid][k]; }
      float m = s * (1.f / 128.f);
      float v = qq * (1.f / 128.f) - m * m;
      mrs[tid][0] = m; mrs[tid][1] = rsqrtf(v + EPS_);
    }
    __syncthreads();
    {
      int row = tid >> 3, seg = tid & 7;
      float m = mrs[row][0], rs = mrs[row][1];
      for (int k = 0; k < 16; ++k) {
        int cidx = seg * 16 + k;
        float v = Hnew[row * FSTRIDE + cidx];
        Hnew[row * FSTRIDE + cidx] = (v - m) * rs * g1s[cidx] + be1s[cidx];
      }
    }
    __syncthreads();

    // ---- phase 4: layer-2 projection + 5-wide RNN step (80 dot products)
    if (tid < 16 * H2_) {
      int m = tid / H2_, j = tid % H2_;
      float acc = b2s[j];
      for (int d = 0; d < D_; ++d) acc += Hnew[m * FSTRIDE + d] * Wih2s[j * D_ + d];
      for (int k = 0; k < H2_; ++k) acc += h2[m * H2_ + k] * Whh2s[j * H2_ + k];
      h2n[m * H2_ + j] = tanhf(acc);
    }
    __syncthreads();

    // ---- phase 5: LN over H2=5 + running mean-pool accumulate
    if (tid < 16) {
      float s = 0.f;
      for (int j = 0; j < H2_; ++j) s += h2n[tid * H2_ + j];
      float m = s * (1.f / H2_);
      float v = 0.f;
      for (int j = 0; j < H2_; ++j) { float d = h2n[tid * H2_ + j] - m; v += d * d; }
      float rs = rsqrtf(v * (1.f / H2_) + EPS_);
      for (int j = 0; j < H2_; ++j) {
        float z = (h2n[tid * H2_ + j] - m) * rs * g2s[j] + be2s[j];
        pooled[tid * H2_ + j] += z;
        h2[tid * H2_ + j] = h2n[tid * H2_ + j];
      }
    }
    __syncthreads();
  }

  // ---- epilogue: mean over S + softmax
  if (tid < 16) {
    float p[H2_], mx = -1e30f;
    for (int j = 0; j < H2_; ++j) {
      p[j] = pooled[tid * H2_ + j] * (1.f / S_);
      mx = fmaxf(mx, p[j]);
    }
    float s = 0.f;
    for (int j = 0; j < H2_; ++j) { p[j] = expf(p[j] - mx); s += p[j]; }
    float inv = 1.f / s;
    for (int j = 0; j < H2_; ++j) out[(size_t)(bbase + tid) * H2_ + j] = p[j] * inv;
  }
}

// ---------------- Host launcher ----------------
extern "C" void kernel_launch(void* const* d_in, const int* in_sizes, int n_in,
                              void* d_out, int out_size, void* d_ws, size_t ws_size,
                              hipStream_t stream) {
  (void)in_sizes; (void)n_in; (void)out_size; (void)ws_size;
  const int*   x    = (const int*)  d_in[0];
  const float* emb  = (const float*)d_in[1];
  const float* Wih1 = (const float*)d_in[2];
  const float* bih1 = (const float*)d_in[3];
  const float* Whh1 = (const float*)d_in[4];
  const float* bhh1 = (const float*)d_in[5];
  const float* g1   = (const float*)d_in[6];
  const float* be1  = (const float*)d_in[7];
  const float* Wih2 = (const float*)d_in[8];
  const float* bih2 = (const float*)d_in[9];
  const float* Whh2 = (const float*)d_in[10];
  const float* bhh2 = (const float*)d_in[11];
  const float* g2   = (const float*)d_in[12];
  const float* be2  = (const float*)d_in[13];

  char* ws = (char*)d_ws;
  __bf16* wih1_bf = (__bf16*)(ws + 0);
  __bf16* whh1_bf = (__bf16*)(ws + (64 << 10));
  float*  bias1   = (float*)(ws + (128 << 10));
  float*  bias2   = (float*)(ws + (128 << 10) + 1024);
  __bf16* xproj   = (__bf16*)(ws + (1 << 20));   // [S, B, D] bf16 = 64 MB (L2-resident)

  prep_kernel<<<64, 256, 0, stream>>>(Wih1, Whh1, bih1, bhh1, bih2, bhh2,
                                      wih1_bf, whh1_bf, bias1, bias2);
  xproj_kernel<<<(B_ * S_) / 64, 128, 0, stream>>>(x, emb, wih1_bf, bias1, xproj);
  rnn_kernel<<<B_ / 16, 128, 0, stream>>>(xproj, whh1_bf, g1, be1,
                                          Wih2, bias2, Whh2, g2, be2, (float*)d_out);
}
